// GrokOneMoELayer_46617575031310
// MI455X (gfx1250) — matrix operand order repaired and checked by reading, not compiled, not run on hardware
//
#include <hip/hip_runtime.h>
#include <math.h>

// Problem constants (from reference): B=2, S=1024, D=1024, DF=4096, E=8, K=2
#define T_   2048      // B*S tokens
#define D_   1024
#define DF_  4096
#define E_   8

// ---- CDNA5 WMMA types ----
typedef __attribute__((ext_vector_type(16))) __bf16       v16bf;
typedef __attribute__((ext_vector_type(8)))  float        v8f;
typedef __attribute__((ext_vector_type(4)))  unsigned int v4u;

union FragU {
    v16bf bf;
    v4u   u[2];
};

// CDNA5 async global->LDS copy (ASYNCcnt) with compile-safe fallback.
// Probe round 2: builtin takes generic pointers to int4 (vector_size(16)).
#if __has_builtin(__builtin_amdgcn_global_load_async_to_lds_b128)
#define ASYNC_LDS 1
typedef int v4i_t __attribute__((vector_size(16)));
#else
#define ASYNC_LDS 0
#endif

__device__ __forceinline__ void copy16_g2l(const void* g, void* l) {
#if ASYNC_LDS
    __builtin_amdgcn_global_load_async_to_lds_b128((v4i_t*)g, (v4i_t*)l, 0, 0);
#else
    *(v4u*)l = *(const v4u*)g;
#endif
}

__device__ __forceinline__ void wait_async_copies() {
#if ASYNC_LDS
#if __has_builtin(__builtin_amdgcn_s_wait_asynccnt)
    __builtin_amdgcn_s_wait_asynccnt(0);
#else
    asm volatile("s_wait_asynccnt 0x0" ::: "memory");
#endif
#endif
}

// float -> bf16 round-to-nearest-even
__device__ __forceinline__ unsigned short f2bf(float f) {
    unsigned int u = __float_as_uint(f);
    u += 0x7fffu + ((u >> 16) & 1u);
    return (unsigned short)(u >> 16);
}

__device__ __forceinline__ float gelu_exact(float x) {
    return 0.5f * x * (1.0f + erff(x * 0.70710678118654752440f));
}

// ---------------------------------------------------------------------------
// Zero the dense output region
// ---------------------------------------------------------------------------
__global__ __launch_bounds__(256) void zero_out_kernel(float* __restrict__ out, int n) {
    int i = blockIdx.x * 256 + threadIdx.x;
    if (i < n) out[i] = 0.0f;
}

// ---------------------------------------------------------------------------
// Convert x (fp32) -> bf16 workspace
// ---------------------------------------------------------------------------
__global__ __launch_bounds__(256) void convert_x_kernel(const float* __restrict__ x,
                                                        unsigned short* __restrict__ xbf,
                                                        int n) {
    int i = blockIdx.x * 256 + threadIdx.x;
    if (i < n) xbf[i] = f2bf(x[i]);
}

// ---------------------------------------------------------------------------
// Transpose + convert one weight matrix: src fp32 [R,C] -> dst bf16 [C,R].
// 32x32 tiles staged through LDS so both global accesses are coalesced.
// ---------------------------------------------------------------------------
__global__ __launch_bounds__(256) void transpose_convert_kernel(
    const float* __restrict__ src, unsigned short* __restrict__ dst, int R, int C) {
    __shared__ unsigned short tile[32][33];
    const int r0 = blockIdx.y * 32;
    const int c0 = blockIdx.x * 32;
    const int tr  = threadIdx.x >> 3;   // 0..31
    const int tc4 = threadIdx.x & 7;    // 0..7 (x4 floats)

    float4 v = *(const float4*)(src + (size_t)(r0 + tr) * C + c0 + tc4 * 4);
    tile[tr][tc4 * 4 + 0] = f2bf(v.x);
    tile[tr][tc4 * 4 + 1] = f2bf(v.y);
    tile[tr][tc4 * 4 + 2] = f2bf(v.z);
    tile[tr][tc4 * 4 + 3] = f2bf(v.w);
    __syncthreads();

    const int cc   = threadIdx.x >> 3;  // output row (over C)
    const int rseg = threadIdx.x & 7;   // x4 elements along R
    ushort4 o;
    o.x = tile[rseg * 4 + 0][cc];
    o.y = tile[rseg * 4 + 1][cc];
    o.z = tile[rseg * 4 + 2][cc];
    o.w = tile[rseg * 4 + 3][cc];
    *(ushort4*)(dst + (size_t)(c0 + cc) * R + r0 + rseg * 4) = o;
}

// ---------------------------------------------------------------------------
// Router: one wave32 per token. logits = x_row @ gate_w, softmax, top-2,
// renormalize; writes router_probs (output tail) and combine weights (ws).
// ---------------------------------------------------------------------------
__global__ __launch_bounds__(256) void router_kernel(const float* __restrict__ x,
                                                     const float* __restrict__ gate_w,
                                                     float* __restrict__ probs_out,
                                                     float* __restrict__ combine) {
    const int token = blockIdx.x * 8 + (threadIdx.x >> 5);
    const int lane  = threadIdx.x & 31;
    const float* xr = x + (size_t)token * D_;

    float acc[E_];
#pragma unroll
    for (int e = 0; e < E_; ++e) acc[e] = 0.0f;

    for (int d = lane; d < D_; d += 32) {
        float xv = xr[d];
        const float4* gw = (const float4*)(gate_w + (size_t)d * E_);
        float4 g0 = gw[0];
        float4 g1 = gw[1];
        acc[0] += xv * g0.x; acc[1] += xv * g0.y;
        acc[2] += xv * g0.z; acc[3] += xv * g0.w;
        acc[4] += xv * g1.x; acc[5] += xv * g1.y;
        acc[6] += xv * g1.z; acc[7] += xv * g1.w;
    }
#pragma unroll
    for (int off = 16; off > 0; off >>= 1) {
#pragma unroll
        for (int e = 0; e < E_; ++e) acc[e] += __shfl_xor(acc[e], off, 32);
    }

    if (lane == 0) {
        float mx = acc[0];
#pragma unroll
        for (int e = 1; e < E_; ++e) mx = fmaxf(mx, acc[e]);
        float p[E_];
        float s = 0.0f;
#pragma unroll
        for (int e = 0; e < E_; ++e) { p[e] = __expf(acc[e] - mx); s += p[e]; }
        float inv = 1.0f / s;
#pragma unroll
        for (int e = 0; e < E_; ++e) {
            p[e] *= inv;
            probs_out[(size_t)token * E_ + e] = p[e];
        }
        int i1 = 0;
#pragma unroll
        for (int e = 1; e < E_; ++e) if (p[e] > p[i1]) i1 = e;
        int i2 = (i1 == 0) ? 1 : 0;
#pragma unroll
        for (int e = 0; e < E_; ++e) if (e != i1 && p[e] > p[i2]) i2 = e;
        float inv2 = 1.0f / (p[i1] + p[i2]);
#pragma unroll
        for (int e = 0; e < E_; ++e) {
            float c = (e == i1) ? p[i1] * inv2 : ((e == i2) ? p[i2] * inv2 : 0.0f);
            combine[(size_t)token * E_ + e] = c;
        }
    }
}

// ---------------------------------------------------------------------------
// FFN stage 1 (per expert): H[t,f] = combine[t,e] * gelu(x@Win)[t,f] * (x@Wv)[t,f]
// M=2048, N=4096 (DF), K=1024 (D). Block 64x128, 8 waves, wave tile 32x32.
// Weights arrive bf16 PRE-TRANSPOSED: wT[N][K]. Double-buffered LDS with
// CDNA5 async global->LDS b128 copies overlapping the WMMA pipe.
// ---------------------------------------------------------------------------
__global__ __launch_bounds__(256) void ffn1_kernel(const unsigned short* __restrict__ xbf,
                                                   const unsigned short* __restrict__ wInT,
                                                   const unsigned short* __restrict__ wVT,
                                                   const float* __restrict__ combine,
                                                   unsigned short* __restrict__ H,
                                                   int expert) {
    __shared__ unsigned short sA[2][64 * 40];
    __shared__ unsigned short sBi[2][128 * 40];
    __shared__ unsigned short sBv[2][128 * 40];

    const int tid   = threadIdx.x;
    const int wid   = tid >> 5;
    const int lane  = tid & 31;
    const int lh    = lane >> 4;
    const int lrow  = lane & 15;
    const int waveM = wid >> 2;
    const int waveN = wid & 3;
    const int m0 = blockIdx.y * 64;
    const int n0 = blockIdx.x * 128;

    v8f accI[2][2], accV[2][2];
#pragma unroll
    for (int a = 0; a < 2; ++a)
#pragma unroll
        for (int b = 0; b < 2; ++b) { accI[a][b] = (v8f){}; accV[a][b] = (v8f){}; }

    const int arow = tid >> 2;      // 0..63
    const int aseg = tid & 3;       // x8 bf16

    auto load_tiles = [&](int buf, int kb) {
        copy16_g2l(xbf + (size_t)(m0 + arow) * D_ + kb + aseg * 8,
                   &sA[buf][arow * 40 + aseg * 8]);
#pragma unroll
        for (int p = 0; p < 2; ++p) {
            int s   = tid + p * 256;   // 0..511
            int n   = s >> 2;          // 0..127
            int seg = s & 3;           // x8 bf16
            size_t off = (size_t)(n0 + n) * D_ + kb + seg * 8;
            copy16_g2l(wInT + off, &sBi[buf][n * 40 + seg * 8]);
            copy16_g2l(wVT  + off, &sBv[buf][n * 40 + seg * 8]);
        }
    };

    load_tiles(0, 0);
    const int NIT = D_ / 32;
    for (int it = 0; it < NIT; ++it) {
        const int cur = it & 1;
        wait_async_copies();
        __syncthreads();
        if (it + 1 < NIT) load_tiles(cur ^ 1, (it + 1) * 32);

        FragU a[2];
#pragma unroll
        for (int tm = 0; tm < 2; ++tm) {
            int r = waveM * 32 + tm * 16 + lrow;
            a[tm].u[0] = *(const v4u*)(&sA[cur][r * 40 + 16 * lh]);
            a[tm].u[1] = *(const v4u*)(&sA[cur][r * 40 + 16 * lh + 8]);
        }
#pragma unroll
        for (int tn = 0; tn < 2; ++tn) {
            int c = waveN * 32 + tn * 16 + lrow;
            FragU bi, bv;
            bi.u[0] = *(const v4u*)(&sBi[cur][c * 40 + 16 * lh]);
            bi.u[1] = *(const v4u*)(&sBi[cur][c * 40 + 16 * lh + 8]);
            bv.u[0] = *(const v4u*)(&sBv[cur][c * 40 + 16 * lh]);
            bv.u[1] = *(const v4u*)(&sBv[cur][c * 40 + 16 * lh + 8]);
#pragma unroll
            for (int tm = 0; tm < 2; ++tm) {
                accI[tm][tn] = __builtin_amdgcn_wmma_f32_16x16x32_bf16(
                    false, a[tm].bf, false, bi.bf, (short)0, accI[tm][tn], false, false);
                accV[tm][tn] = __builtin_amdgcn_wmma_f32_16x16x32_bf16(
                    false, a[tm].bf, false, bv.bf, (short)0, accV[tm][tn], false, false);
            }
        }
        __syncthreads();
    }

#pragma unroll
    for (int tm = 0; tm < 2; ++tm) {
#pragma unroll
        for (int tn = 0; tn < 2; ++tn) {
#pragma unroll
            for (int j = 0; j < 8; ++j) {
                int row = m0 + waveM * 32 + tm * 16 + 8 * lh + j;  // C/D: M=j+8*half
                int col = n0 + waveN * 32 + tn * 16 + lrow;        // N=lane&15
                float hv = gelu_exact(accI[tm][tn][j]) * accV[tm][tn][j];
                hv *= combine[(size_t)row * E_ + expert];
                H[(size_t)row * DF_ + col] = f2bf(hv);
            }
        }
    }
}

// ---------------------------------------------------------------------------
// FFN stage 2 (per expert): out[t,d] += (H @ Wout)[t,d]
// M=2048, N=1024 (D), K=4096 (DF). Wout pre-transposed bf16 [D][DF].
// ---------------------------------------------------------------------------
__global__ __launch_bounds__(256) void ffn2_kernel(const unsigned short* __restrict__ H,
                                                   const unsigned short* __restrict__ wOutT,
                                                   float* __restrict__ out) {
    __shared__ unsigned short sA[2][64 * 40];
    __shared__ unsigned short sB[2][128 * 40];

    const int tid   = threadIdx.x;
    const int wid   = tid >> 5;
    const int lane  = tid & 31;
    const int lh    = lane >> 4;
    const int lrow  = lane & 15;
    const int waveM = wid >> 2;
    const int waveN = wid & 3;
    const int m0 = blockIdx.y * 64;
    const int n0 = blockIdx.x * 128;

    v8f acc[2][2];
#pragma unroll
    for (int a = 0; a < 2; ++a)
#pragma unroll
        for (int b = 0; b < 2; ++b) acc[a][b] = (v8f){};

    const int arow = tid >> 2;
    const int aseg = tid & 3;

    auto load_tiles = [&](int buf, int kb) {
        copy16_g2l(H + (size_t)(m0 + arow) * DF_ + kb + aseg * 8,
                   &sA[buf][arow * 40 + aseg * 8]);
#pragma unroll
        for (int p = 0; p < 2; ++p) {
            int s   = tid + p * 256;
            int n   = s >> 2;
            int seg = s & 3;
            copy16_g2l(wOutT + (size_t)(n0 + n) * DF_ + kb + seg * 8,
                       &sB[buf][n * 40 + seg * 8]);
        }
    };

    load_tiles(0, 0);
    const int NIT = DF_ / 32;
    for (int it = 0; it < NIT; ++it) {
        const int cur = it & 1;
        wait_async_copies();
        __syncthreads();
        if (it + 1 < NIT) load_tiles(cur ^ 1, (it + 1) * 32);

        FragU a[2];
#pragma unroll
        for (int tm = 0; tm < 2; ++tm) {
            int r = waveM * 32 + tm * 16 + lrow;
            a[tm].u[0] = *(const v4u*)(&sA[cur][r * 40 + 16 * lh]);
            a[tm].u[1] = *(const v4u*)(&sA[cur][r * 40 + 16 * lh + 8]);
        }
#pragma unroll
        for (int tn = 0; tn < 2; ++tn) {
            int c = waveN * 32 + tn * 16 + lrow;
            FragU b;
            b.u[0] = *(const v4u*)(&sB[cur][c * 40 + 16 * lh]);
            b.u[1] = *(const v4u*)(&sB[cur][c * 40 + 16 * lh + 8]);
#pragma unroll
            for (int tm = 0; tm < 2; ++tm) {
                acc[tm][tn] = __builtin_amdgcn_wmma_f32_16x16x32_bf16(
                    false, a[tm].bf, false, b.bf, (short)0, acc[tm][tn], false, false);
            }
        }
        __syncthreads();
    }

#pragma unroll
    for (int tm = 0; tm < 2; ++tm) {
#pragma unroll
        for (int tn = 0; tn < 2; ++tn) {
#pragma unroll
            for (int j = 0; j < 8; ++j) {
                int row = m0 + waveM * 32 + tm * 16 + 8 * lh + j;
                int col = n0 + waveN * 32 + tn * 16 + lrow;
                size_t idx = (size_t)row * D_ + col;
                out[idx] += acc[tm][tn][j];   // expert kernels serialize on stream
            }
        }
    }
}

// ---------------------------------------------------------------------------
extern "C" void kernel_launch(void* const* d_in, const int* in_sizes, int n_in,
                              void* d_out, int out_size, void* d_ws, size_t ws_size,
                              hipStream_t stream) {
    (void)in_sizes; (void)n_in; (void)out_size; (void)ws_size;

    const float* x      = (const float*)d_in[0];
    const float* gate_w = (const float*)d_in[1];
    const float* w_in   = (const float*)d_in[2];
    const float* w_v    = (const float*)d_in[3];
    const float* w_out  = (const float*)d_in[4];

    float* out   = (float*)d_out;                  // [T_, D_]
    float* probs = out + (size_t)T_ * D_;          // [T_, E_] output tail

    // ws: combine 64KB | x_bf16 4MB | H 16MB | wInT 8MB | wVT 8MB | wOutT 8MB
    char* ws = (char*)d_ws;
    float*          combine = (float*)ws;
    unsigned short* xbf     = (unsigned short*)(ws + 65536);
    unsigned short* Hbuf    = (unsigned short*)(ws + 65536 + (size_t)T_ * D_ * 2);
    unsigned short* wInT    = (unsigned short*)(ws + 65536 + (size_t)T_ * D_ * 2 + (size_t)T_ * DF_ * 2);
    unsigned short* wVT     = wInT + (size_t)D_ * DF_;
    unsigned short* wOutT   = wVT  + (size_t)D_ * DF_;

    const int nOut = T_ * D_;
    zero_out_kernel<<<(nOut + 255) / 256, 256, 0, stream>>>(out, nOut);
    convert_x_kernel<<<(nOut + 255) / 256, 256, 0, stream>>>(x, xbf, nOut);
    router_kernel<<<T_ / 8, 256, 0, stream>>>(x, gate_w, probs, combine);

    dim3 gT1(DF_ / 32, D_ / 32);   // w_in/w_v: [D_,DF_] -> [DF_,D_]
    dim3 gT2(D_ / 32, DF_ / 32);   // w_out:   [DF_,D_] -> [D_,DF_]
    dim3 g1(DF_ / 128, T_ / 64);
    dim3 g2(D_ / 128,  T_ / 64);

    for (int e = 0; e < E_; ++e) {
        transpose_convert_kernel<<<gT1, 256, 0, stream>>>(w_in + (size_t)e * D_ * DF_, wInT, D_, DF_);
        transpose_convert_kernel<<<gT1, 256, 0, stream>>>(w_v  + (size_t)e * D_ * DF_, wVT,  D_, DF_);
        transpose_convert_kernel<<<gT2, 256, 0, stream>>>(w_out + (size_t)e * DF_ * D_, wOutT, DF_, D_);
        ffn1_kernel<<<g1, 256, 0, stream>>>(xbf, wInT, wVT, combine, Hbuf, e);
        ffn2_kernel<<<g2, 256, 0, stream>>>(Hbuf, wOutT, out);
    }
}
